// EnhancedCellLevelGNN_40432822124920
// MI455X (gfx1250) — matrix operand order repaired and checked by reading, not compiled, not run on hardware
//
#include <hip/hip_runtime.h>
#include <hip/hip_bf16.h>
#include <cstdint>

// ---------------------------------------------------------------------------
// Types for CDNA5 wave32 WMMA (gfx1250): v_wmma_f32_16x16x32_bf16
// ---------------------------------------------------------------------------
typedef __hip_bfloat16 bf16s;                                   // storage
typedef __attribute__((ext_vector_type(16))) __bf16 v16bf;      // A/B fragment (8 VGPRs)
typedef __attribute__((ext_vector_type(8)))  float  v8f;        // C/D fragment (8 VGPRs)

union FragBF { v16bf v; uint4 q[2]; };
static_assert(sizeof(v16bf) == 32, "bf16 fragment must be 32 bytes");

__device__ __forceinline__ float bf2f(bf16s x) { return __bfloat162float(x); }
__device__ __forceinline__ bf16s f2bf(float x) { return __float2bfloat16(x); }

// ---------------------------------------------------------------------------
// WMMA GEMM:  C[M,N] = A[M,K](bf16,row-major) x Bt[N,K](bf16, W pre-transposed)
// epilogue: +bias (optional), +=C (ACC), relu (RELU), bf16 or f32 store.
// Block: 256 threads = 8 waves; each wave -> 16 x (16*NT) output strip.
// M multiple of 16, N multiple of 16*NT, K multiple of 64 (true for all calls).
// K-loop is ping-pong double-buffered so fragment loads of block i+1 overlap
// the WMMAs of block i (partial s_wait_loadcnt instead of 0).
// ---------------------------------------------------------------------------
template<int NT, int RELU, int ACC, int OUTBF>
__global__ __launch_bounds__(256) void k_gemm(const bf16s* __restrict__ Abuf,
                                              const bf16s* __restrict__ Bt,
                                              const float* __restrict__ bias,
                                              void* __restrict__ Cout,
                                              int M, int N, int K)
{
    const int lane = threadIdx.x & 31;
    const int wv   = threadIdx.x >> 5;
    const int m0   = blockIdx.y * 128 + wv * 16;
    if (m0 >= M) return;                       // wave-uniform: EXEC stays all-1s
    const int n0   = blockIdx.x * (16 * NT);

    // A 16x32 bf16 layout: lane half 0 -> K {0..15}; half 1 -> {16..31}
    const int arow = m0 + (lane & 15);
    const int ah   = lane >> 4;
    // B 32x16 bf16 layout: lane%16 = column N, lane half selects K {0..15}/{16..31}
    const int bcol = lane & 15;
    const int bh   = lane >> 4;

    v8f acc[NT];
#pragma unroll
    for (int j = 0; j < NT; ++j)
#pragma unroll
        for (int i = 0; i < 8; ++i) acc[j][i] = 0.0f;

    const size_t abase = (size_t)arow * K;

    auto load_frags = [&](FragBF& af, FragBF (&bfr)[NT], int kb) {
        af.q[0] = *(const uint4*)(Abuf + abase + kb + ah * 16);
        af.q[1] = *(const uint4*)(Abuf + abase + kb + ah * 16 + 8);
#pragma unroll
        for (int j = 0; j < NT; ++j) {
            const bf16s* bp = Bt + (size_t)(n0 + j * 16 + bcol) * K + kb + bh * 16;
            bfr[j].q[0] = *(const uint4*)bp;
            bfr[j].q[1] = *(const uint4*)(bp + 8);
        }
    };
    auto mma_all = [&](const FragBF& af, const FragBF (&bfr)[NT]) {
#pragma unroll
        for (int j = 0; j < NT; ++j)
            acc[j] = __builtin_amdgcn_wmma_f32_16x16x32_bf16(
                false, af.v, false, bfr[j].v, (short)0, acc[j], false, false);
    };

    FragBF a0, a1;
    FragBF b0[NT], b1[NT];
    load_frags(a0, b0, 0);
    int kb = 0;
    for (; kb + 64 < K; kb += 64) {            // K % 64 == 0 for all call sites
        load_frags(a1, b1, kb + 32);
        mma_all(a0, b0);
        load_frags(a0, b0, kb + 64);
        mma_all(a1, b1);
    }
    load_frags(a1, b1, kb + 32);
    mma_all(a0, b0);
    mma_all(a1, b1);

    // D layout: lanes 0-15 -> rows m0+g, lanes 16-31 -> rows m0+8+g; col = lane%16
    float* Cf = (float*)Cout;
    bf16s* Cb = (bf16s*)Cout;
    const int row0 = m0 + ((lane >> 4) << 3);
#pragma unroll
    for (int j = 0; j < NT; ++j) {
        const int n  = n0 + j * 16 + (lane & 15);
        const float bv = bias ? bias[n] : 0.0f;
#pragma unroll
        for (int g = 0; g < 8; ++g) {
            const size_t idx = (size_t)(row0 + g) * N + n;
            float v = acc[j][g] + bv;
            if (ACC)  v += Cf[idx];
            if (RELU) v = fmaxf(v, 0.0f);
            if (OUTBF) Cb[idx] = f2bf(v); else Cf[idx] = v;
        }
    }
}

// ---------------------------------------------------------------------------
// LayerNorm(+ReLU) -> bf16. One wave per row (wave32 shuffle reduction).
// ---------------------------------------------------------------------------
__global__ __launch_bounds__(256) void k_ln(const float* __restrict__ X,
                                            const float* __restrict__ g,
                                            const float* __restrict__ b,
                                            bf16s* __restrict__ out,
                                            int rows, int D)
{
    const int lane = threadIdx.x & 31;
    const int row  = blockIdx.x * 8 + (threadIdx.x >> 5);
    if (row >= rows) return;
    const float* x = X + (size_t)row * D;
    float s = 0.f, ss = 0.f;
    for (int i = lane; i < D; i += 32) { float v = x[i]; s += v; ss += v * v; }
    for (int off = 16; off > 0; off >>= 1) {
        s  += __shfl_xor(s,  off, 32);
        ss += __shfl_xor(ss, off, 32);
    }
    const float mean = s / D;
    const float var  = fmaxf(ss / D - mean * mean, 0.f);
    const float rstd = rsqrtf(var + 1e-5f);
    bf16s* o = out + (size_t)row * D;
    for (int i = lane; i < D; i += 32) {
        float v = (x[i] - mean) * rstd * g[i] + b[i];
        o[i] = f2bf(fmaxf(v, 0.f));
    }
}

// ---------------------------------------------------------------------------
// Elementwise helpers
// ---------------------------------------------------------------------------
__global__ void k_cast(const float* __restrict__ s, bf16s* __restrict__ d, int n) {
    int i = blockIdx.x * 256 + threadIdx.x;
    if (i < n) d[i] = f2bf(s[i]);
}

__global__ void k_resid(const float* __restrict__ a, const float* __restrict__ b,
                        bf16s* __restrict__ o, int n) {               // o = bf16(a + b)
    int i = blockIdx.x * 256 + threadIdx.x;
    if (i < n) o[i] = f2bf(a[i] + b[i]);
}

__global__ void k_transpose(const float* __restrict__ W, bf16s* __restrict__ Wt,
                            int K, int N) {                           // W[K,N] -> Wt[N,K]
    int i = blockIdx.x * 256 + threadIdx.x;
    if (i >= K * N) return;
    int n = i / K, k = i - n * K;
    Wt[i] = f2bf(W[(size_t)k * N + n]);
}

// ---------------------------------------------------------------------------
// SAGE aggregation (mean over incoming edges)
// ---------------------------------------------------------------------------
__global__ void k_deg(const int* __restrict__ dst, float* __restrict__ deg, int E) {
    int i = blockIdx.x * 256 + threadIdx.x;
    if (i < E) atomicAdd(&deg[dst[i]], 1.0f);
}

__global__ __launch_bounds__(256) void k_scatter_rows(const bf16s* __restrict__ h,
                                                      const int* __restrict__ src,
                                                      const int* __restrict__ dst,
                                                      float* __restrict__ agg) {
    const int e = blockIdx.x, t = threadIdx.x;             // 256 features per row
    atomicAdd(&agg[(size_t)dst[e] * 256 + t], bf2f(h[(size_t)src[e] * 256 + t]));
}

__global__ void k_divdeg(const float* __restrict__ agg, const float* __restrict__ deg,
                         bf16s* __restrict__ out, int total) {
    int i = blockIdx.x * 256 + threadIdx.x;
    if (i >= total) return;
    out[i] = f2bf(agg[i] / fmaxf(deg[i >> 8], 1.0f));
}

// ---------------------------------------------------------------------------
// GATv2 (heads=4, C=256, self-loops appended as edges E..E+N-1)
// ---------------------------------------------------------------------------
__device__ __forceinline__ void edge_sd(int e, int E, const int* src, const int* dst,
                                        int& s, int& d) {
    if (e < E) { s = src[e]; d = dst[e]; } else { s = d = e - E; }
}

// e[edge,head] = sum_c lrelu(xl[s]+xr[d]) * att ; atomic segment-max via monotonic uint
__global__ __launch_bounds__(256) void k_gat_edge(const bf16s* __restrict__ xl,
                                                  const bf16s* __restrict__ xr,
                                                  const int* __restrict__ src,
                                                  const int* __restrict__ dst,
                                                  const float* __restrict__ att,
                                                  float* __restrict__ ebuf,
                                                  unsigned* __restrict__ mmax,
                                                  int E, int N) {
    const int lane = threadIdx.x & 31;
    const int wid  = blockIdx.x * 8 + (threadIdx.x >> 5);
    if (wid >= (E + N) * 4) return;
    const int edge = wid >> 2, head = wid & 3;
    int s, d; edge_sd(edge, E, src, dst, s, d);
    const bf16s* pl = xl + (size_t)s * 1024 + head * 256;
    const bf16s* pr = xr + (size_t)d * 1024 + head * 256;
    const float* pa = att + head * 256;
    float e = 0.f;
    for (int c = lane; c < 256; c += 32) {
        float v = bf2f(pl[c]) + bf2f(pr[c]);
        v = v > 0.f ? v : 0.2f * v;                 // leaky_relu(0.2)
        e += v * pa[c];
    }
    for (int off = 16; off > 0; off >>= 1) e += __shfl_xor(e, off, 32);
    if (lane == 0) {
        ebuf[wid] = e;
        unsigned u = __float_as_uint(e);
        unsigned mapped = (e >= 0.f) ? (u | 0x80000000u) : ~u;   // monotonic map
        atomicMax(&mmax[(size_t)d * 4 + head], mapped);
    }
}

__global__ void k_gat_exp(const float* __restrict__ ebuf, const unsigned* __restrict__ mmax,
                          const int* __restrict__ src, const int* __restrict__ dst,
                          float* __restrict__ exbuf, float* __restrict__ denom,
                          int E, int N) {
    int i = blockIdx.x * 256 + threadIdx.x;
    if (i >= (E + N) * 4) return;
    const int edge = i >> 2, head = i & 3;
    int s, d; edge_sd(edge, E, src, dst, s, d);
    unsigned u = mmax[(size_t)d * 4 + head];
    float m = (u & 0x80000000u) ? __uint_as_float(u & 0x7FFFFFFFu) : __uint_as_float(~u);
    float ex = __expf(ebuf[i] - m);
    exbuf[i] = ex;
    atomicAdd(&denom[(size_t)d * 4 + head], ex);
}

__global__ __launch_bounds__(256) void k_gat_scatter(const float* __restrict__ exbuf,
                                                     const float* __restrict__ denom,
                                                     const bf16s* __restrict__ xl,
                                                     const int* __restrict__ src,
                                                     const int* __restrict__ dst,
                                                     float* __restrict__ outacc,
                                                     int E, int N) {
    const int e = blockIdx.x, t = threadIdx.x;
    int s, d; edge_sd(e, E, src, dst, s, d);
#pragma unroll
    for (int h = 0; h < 4; ++h) {
        float alpha = exbuf[(size_t)e * 4 + h] / denom[(size_t)d * 4 + h];
        atomicAdd(&outacc[(size_t)d * 1024 + h * 256 + t],
                  alpha * bf2f(xl[(size_t)s * 1024 + h * 256 + t]));
    }
}

__global__ void k_gat_fin(const float* __restrict__ acc, const float* __restrict__ bias,
                          float* __restrict__ h1f, bf16s* __restrict__ h1b, int total) {
    int i = blockIdx.x * 256 + threadIdx.x;
    if (i >= total) return;
    const int n = i >> 8, c = i & 255;
    const float* a = acc + (size_t)n * 1024 + c;
    float v = (a[0] + a[256] + a[512] + a[768]) * 0.25f + bias[c];
    v = fmaxf(v, 0.f);
    h1f[i] = v;
    h1b[i] = f2bf(v);
}

// comb[row, 0:256] = h2[cell_patches[row]]; comb[row, 256:384] = c2[row]
__global__ void k_concat(const bf16s* __restrict__ h2b, const bf16s* __restrict__ c2b,
                         const int* __restrict__ patch, bf16s* __restrict__ comb, int NC) {
    int i = blockIdx.x * 256 + threadIdx.x;
    if (i >= NC * 384) return;
    int row = i / 384, col = i - row * 384;
    comb[i] = (col < 256) ? h2b[(size_t)patch[row] * 256 + col]
                          : c2b[(size_t)row * 128 + (col - 256)];
}

// ---------------------------------------------------------------------------
// Launcher
// ---------------------------------------------------------------------------
extern "C" void kernel_launch(void* const* d_in, const int* in_sizes, int n_in,
                              void* d_out, int out_size, void* d_ws, size_t ws_size,
                              hipStream_t stream)
{
    const float* x    = (const float*)d_in[0];
    const int*   eidx = (const int*)d_in[1];
    const float* cf   = (const float*)d_in[2];
    const int*   cpat = (const int*)d_in[3];
    const int Nn = in_sizes[0] / 1024;
    const int E  = in_sizes[1] / 2;
    const int NC = in_sizes[2] / 64;
    const int* src = eidx;
    const int* dst = eidx + E;

    // params flattened in sorted-key order (Python dict insertion = sorted names)
    const float *ce_W1=(const float*)d_in[4],  *ce_W2=(const float*)d_in[5],
                *ce_b1=(const float*)d_in[6],  *ce_b2=(const float*)d_in[7],
                *ce_be1=(const float*)d_in[8], *ce_be2=(const float*)d_in[9],
                *ce_g1=(const float*)d_in[10], *ce_g2=(const float*)d_in[11],
                *cm_W=(const float*)d_in[12],  *cm_b=(const float*)d_in[13],
                *cm_be=(const float*)d_in[14], *cm_g=(const float*)d_in[15],
                *g_Wl=(const float*)d_in[16],  *g_Wr=(const float*)d_in[17],
                *g_att=(const float*)d_in[18], *g_bias=(const float*)d_in[19],
                *g_bl=(const float*)d_in[20],  *g_br=(const float*)d_in[21],
                *o_W1=(const float*)d_in[22],  *o_W2=(const float*)d_in[23],
                *o_W3=(const float*)d_in[24],  *o_b1=(const float*)d_in[25],
                *o_b2=(const float*)d_in[26],  *o_b3=(const float*)d_in[27],
                *pe_W1=(const float*)d_in[28], *pe_W2=(const float*)d_in[29],
                *pe_b1=(const float*)d_in[30], *pe_b2=(const float*)d_in[31],
                *pe_be1=(const float*)d_in[32],*pe_be2=(const float*)d_in[33],
                *pe_g1=(const float*)d_in[34], *pe_g2=(const float*)d_in[35],
                *s0_Wl=(const float*)d_in[36], *s0_Wr=(const float*)d_in[37],
                *s0_bl=(const float*)d_in[38], *s2_Wl=(const float*)d_in[39],
                *s2_Wr=(const float*)d_in[40], *s2_bl=(const float*)d_in[41];

    // ---- workspace arena with lifetime-based slot reuse ----
    char* base = (char*)d_ws;
    size_t off = 0;
    auto arena = [&](size_t bytes) -> char* {
        char* r = base + off; off += (bytes + 255) & ~(size_t)255; return r;
    };

    // U1: xb|t1 (PE) -> xl|xr (GAT, bf16) -> combb|cellb (cell phase)
    char* U1 = arena((size_t)Nn * 1024 * 2 + (size_t)Nn * 512 * 4);
    bf16s* xb    = (bf16s*)U1;
    float* t1    = (float*)(U1 + (size_t)Nn * 1024 * 2);
    bf16s* xl    = (bf16s*)U1;
    bf16s* xr    = (bf16s*)(U1 + (size_t)Nn * 1024 * 2);
    bf16s* combb = (bf16s*)U1;
    bf16s* cellb = (bf16s*)(U1 + (size_t)NC * 384 * 2);
    // U2: hb1|t2|hb|h0b -> c1 -> comblnb
    char* U2 = arena((size_t)NC * 256 * 4);
    bf16s* hb1 = (bf16s*)U2;
    float* t2  = (float*)(U2 + (size_t)Nn * 512 * 2);
    bf16s* hb  = (bf16s*)(U2 + (size_t)Nn * 512 * 2 + (size_t)Nn * 256 * 4);
    bf16s* h0b = (bf16s*)(U2 + (size_t)Nn * 512 * 2 + (size_t)Nn * 256 * 4
                             + (size_t)Nn * 256 * 2);
    float* c1       = (float*)U2;
    bf16s* comblnb  = (bf16s*)U2;
    // U3: gatout -> combf -> o1f (all f32, same footprint)
    char* U3 = arena((size_t)NC * 256 * 4);
    float* gatout = (float*)U3;
    float* combf  = (float*)U3;
    float* o1f    = (float*)U3;
    // U4: c2f|c2b -> o2f|o2b
    char* U4 = arena((size_t)NC * 128 * 6);
    float* c2f = (float*)U4;  bf16s* c2b = (bf16s*)(U4 + (size_t)NC * 128 * 4);
    float* o2f = (float*)U4;  bf16s* o2b = c2b;
    // U5: c1b | o1b
    char* U5 = arena((size_t)NC * 256 * 4);
    bf16s* c1b = (bf16s*)U5;
    bf16s* o1b = (bf16s*)(U5 + (size_t)NC * 256 * 2);

    float*    agg   = (float*)arena((size_t)Nn * 256 * 4);
    bf16s*    aggb  = (bf16s*)arena((size_t)Nn * 256 * 2);
    float*    s0out = (float*)arena((size_t)Nn * 256 * 4);
    float*    deg   = (float*)arena((size_t)Nn * 4);
    float*    ebuf  = (float*)arena((size_t)(E + Nn) * 4 * 4);
    float*    exbuf = (float*)arena((size_t)(E + Nn) * 4 * 4);
    unsigned* mmax  = (unsigned*)arena((size_t)Nn * 16);
    float*    denom = (float*)arena((size_t)Nn * 16);
    float*    h1f   = (float*)arena((size_t)Nn * 256 * 4);
    bf16s*    h1b   = (bf16s*)arena((size_t)Nn * 256 * 2);
    bf16s*    h2b   = (bf16s*)arena((size_t)Nn * 256 * 2);
    // bf16-transposed weights Wt[N,K]
    bf16s* peW1t = (bf16s*)arena(512 * 1024 * 2);
    bf16s* peW2t = (bf16s*)arena(256 * 512 * 2);
    bf16s* s0Wlt = (bf16s*)arena(256 * 256 * 2);
    bf16s* s0Wrt = (bf16s*)arena(256 * 256 * 2);
    bf16s* gWlt  = (bf16s*)arena(1024 * 256 * 2);
    bf16s* gWrt  = (bf16s*)arena(1024 * 256 * 2);
    bf16s* s2Wlt = (bf16s*)arena(256 * 256 * 2);
    bf16s* s2Wrt = (bf16s*)arena(256 * 256 * 2);
    bf16s* ceW1t = (bf16s*)arena(256 * 64 * 2);
    bf16s* ceW2t = (bf16s*)arena(128 * 256 * 2);
    bf16s* cmWt  = (bf16s*)arena(256 * 384 * 2);
    bf16s* oW1t  = (bf16s*)arena(256 * 256 * 2);
    bf16s* oW2t  = (bf16s*)arena(128 * 256 * 2);
    bf16s* oW3t  = (bf16s*)arena(32 * 128 * 2);

    const dim3 blk(256);
    auto cdiv = [](int a, int b) { return (a + b - 1) / b; };

#define TR(W, Wt, Kd, Nd) \
    k_transpose<<<cdiv((Kd) * (Nd), 256), blk, 0, stream>>>(W, Wt, Kd, Nd)
#define GEMM(NT, R, AC, OB, Ab, Bt, bias, C, M, Nd, Kd) \
    k_gemm<NT, R, AC, OB><<<dim3((Nd) / (16 * (NT)), cdiv(M, 128)), blk, 0, stream>>>( \
        Ab, Bt, bias, C, M, Nd, Kd)

    // ---- weight prep (bf16 transpose) ----
    TR(pe_W1, peW1t, 1024, 512);  TR(pe_W2, peW2t, 512, 256);
    TR(s0_Wl, s0Wlt, 256, 256);   TR(s0_Wr, s0Wrt, 256, 256);
    TR(g_Wl,  gWlt,  256, 1024);  TR(g_Wr,  gWrt,  256, 1024);
    TR(s2_Wl, s2Wlt, 256, 256);   TR(s2_Wr, s2Wrt, 256, 256);
    TR(ce_W1, ceW1t, 64, 256);    TR(ce_W2, ceW2t, 256, 128);
    TR(cm_W,  cmWt,  384, 256);
    TR(o_W1,  oW1t,  256, 256);   TR(o_W2,  oW2t, 256, 128);
    TR(o_W3,  oW3t,  128, 32);

    // ---- patch embedding MLP ----
    k_cast<<<cdiv(Nn * 1024, 256), blk, 0, stream>>>(x, xb, Nn * 1024);
    GEMM(4, 0, 0, 0, xb, peW1t, pe_b1, t1, Nn, 512, 1024);
    k_ln<<<cdiv(Nn, 8), blk, 0, stream>>>(t1, pe_g1, pe_be1, hb1, Nn, 512);
    GEMM(4, 0, 0, 0, hb1, peW2t, pe_b2, t2, Nn, 256, 512);
    k_ln<<<cdiv(Nn, 8), blk, 0, stream>>>(t2, pe_g2, pe_be2, hb, Nn, 256);

    // ---- SAGE conv 0 ----
    hipMemsetAsync(deg, 0, (size_t)Nn * 4, stream);
    hipMemsetAsync(agg, 0, (size_t)Nn * 256 * 4, stream);
    k_deg<<<cdiv(E, 256), blk, 0, stream>>>(dst, deg, E);
    k_scatter_rows<<<E, blk, 0, stream>>>(hb, src, dst, agg);
    k_divdeg<<<cdiv(Nn * 256, 256), blk, 0, stream>>>(agg, deg, aggb, Nn * 256);
    GEMM(4, 0, 0, 0, aggb, s0Wlt, s0_bl, s0out, Nn, 256, 256);
    GEMM(4, 1, 1, 0, hb, s0Wrt, nullptr, s0out, Nn, 256, 256);   // += , relu
    k_cast<<<cdiv(Nn * 256, 256), blk, 0, stream>>>(s0out, h0b, Nn * 256);

    // ---- GATv2 ----
    GEMM(4, 0, 0, 1, h0b, gWlt, g_bl, xl, Nn, 1024, 256);        // bf16 out
    GEMM(4, 0, 0, 1, h0b, gWrt, g_br, xr, Nn, 1024, 256);        // bf16 out
    hipMemsetAsync(mmax, 0, (size_t)Nn * 16, stream);            // 0 == mapped -inf
    hipMemsetAsync(denom, 0, (size_t)Nn * 16, stream);
    hipMemsetAsync(gatout, 0, (size_t)Nn * 1024 * 4, stream);
    k_gat_edge<<<cdiv((E + Nn) * 4, 8), blk, 0, stream>>>(xl, xr, src, dst, g_att,
                                                          ebuf, mmax, E, Nn);
    k_gat_exp<<<cdiv((E + Nn) * 4, 256), blk, 0, stream>>>(ebuf, mmax, src, dst,
                                                           exbuf, denom, E, Nn);
    k_gat_scatter<<<E + Nn, blk, 0, stream>>>(exbuf, denom, xl, src, dst, gatout, E, Nn);
    k_gat_fin<<<cdiv(Nn * 256, 256), blk, 0, stream>>>(gatout, g_bias, h1f, h1b, Nn * 256);

    // ---- SAGE conv 2 + residual ----
    hipMemsetAsync(agg, 0, (size_t)Nn * 256 * 4, stream);
    k_scatter_rows<<<E, blk, 0, stream>>>(h1b, src, dst, agg);
    k_divdeg<<<cdiv(Nn * 256, 256), blk, 0, stream>>>(agg, deg, aggb, Nn * 256);
    GEMM(4, 0, 0, 0, aggb, s2Wlt, s2_bl, s0out, Nn, 256, 256);
    GEMM(4, 1, 1, 0, h1b, s2Wrt, nullptr, s0out, Nn, 256, 256);  // relu(sage2)
    k_resid<<<cdiv(Nn * 256, 256), blk, 0, stream>>>(s0out, h1f, h2b, Nn * 256);

    // ---- cell embedding MLP ----
    k_cast<<<cdiv(NC * 64, 256), blk, 0, stream>>>(cf, cellb, NC * 64);
    GEMM(4, 0, 0, 0, cellb, ceW1t, ce_b1, c1, NC, 256, 64);
    k_ln<<<cdiv(NC, 8), blk, 0, stream>>>(c1, ce_g1, ce_be1, c1b, NC, 256);
    GEMM(4, 0, 0, 0, c1b, ceW2t, ce_b2, c2f, NC, 128, 256);
    k_ln<<<cdiv(NC, 8), blk, 0, stream>>>(c2f, ce_g2, ce_be2, c2b, NC, 128);

    // ---- combine ----
    k_concat<<<cdiv(NC * 384, 256), blk, 0, stream>>>(h2b, c2b, cpat, combb, NC);
    GEMM(4, 0, 0, 0, combb, cmWt, cm_b, combf, NC, 256, 384);
    k_ln<<<cdiv(NC, 8), blk, 0, stream>>>(combf, cm_g, cm_be, comblnb, NC, 256);

    // ---- output head ----
    GEMM(4, 1, 0, 0, comblnb, oW1t, o_b1, o1f, NC, 256, 256);
    k_cast<<<cdiv(NC * 256, 256), blk, 0, stream>>>(o1f, o1b, NC * 256);
    GEMM(4, 1, 0, 0, o1b, oW2t, o_b2, o2f, NC, 128, 256);
    k_cast<<<cdiv(NC * 128, 256), blk, 0, stream>>>(o2f, o2b, NC * 128);
    GEMM(2, 0, 0, 0, o2b, oW3t, o_b3, (float*)d_out, NC, 32, 128);

#undef TR
#undef GEMM
    (void)n_in; (void)out_size; (void)ws_size;
}
